// CFCCell_38199439131313
// MI455X (gfx1250) — compile-verified, matrix operand order checked by reading, not run on hardware
//
#include <hip/hip_runtime.h>
#include <hip/hip_bf16.h>
#include <math.h>

typedef __attribute__((ext_vector_type(16))) __bf16 v16bf;
typedef __attribute__((ext_vector_type(8)))  float  v8f;
typedef __attribute__((ext_vector_type(4)))  float  f32x4;

#define B_ROWS 131072
#define EPS 1e-5f
#define SLOPE 0.01f

// ---------------------------------------------------------------------------
// K0a: W01 = W0[256,128] @ W1[128,128]  (tiny f32 GEMM, one output per thread)
// ---------------------------------------------------------------------------
__global__ void cfc_w01_kernel(const float* __restrict__ W0,
                               const float* __restrict__ W1,
                               float* __restrict__ W01f) {
    int gid = blockIdx.x * blockDim.x + threadIdx.x;   // 0..32767
    int i = gid >> 7, j = gid & 127;
    float s = 0.f;
#pragma unroll 8
    for (int k = 0; k < 128; ++k) s += W0[i * 128 + k] * W1[k * 128 + j];
    W01f[gid] = s;
}

// ---------------------------------------------------------------------------
// K0b: pack W01 (256x128) and concat heads (128x512) into bf16 WMMA B-fragment
// layout: packed[((kk*T + n)*32 + lane)*16 + e]  with
//   col = n*16 + (lane&15),  k = kk*32 + (e<8 ? (lane>>4)*8+e : 16+(lane>>4)*8+e-8)
// Also zero the global stat accumulators (256 floats).
// ---------------------------------------------------------------------------
__global__ void cfc_pack_kernel(const float* __restrict__ W01f,
                                const float* __restrict__ Wg, const float* __restrict__ Wf,
                                const float* __restrict__ Wh, const float* __restrict__ Wt,
                                __bf16* __restrict__ W01p, __bf16* __restrict__ Hp,
                                float* __restrict__ stats) {
    int gid = blockIdx.x * blockDim.x + threadIdx.x;   // 0..98303
    if (gid < 256) stats[gid] = 0.f;
    if (gid < 32768) {                                  // W01: K=256 -> kk 0..7, 8 tiles
        int e = gid & 15, lane = (gid >> 4) & 31, rest = gid >> 9;
        int n = rest & 7, kk = rest >> 3;
        int l = lane & 15, hi = lane >> 4;
        int k = kk * 32 + ((e < 8) ? (hi * 8 + e) : (16 + hi * 8 + e - 8));
        int col = n * 16 + l;
        W01p[gid] = (__bf16)W01f[k * 128 + col];
    } else {                                            // heads: K=128 -> kk 0..3, 32 tiles
        int q = gid - 32768;                            // 0..65535
        int e = q & 15, lane = (q >> 4) & 31, rest = q >> 9;
        int n = rest & 31, kk = rest >> 5;
        int l = lane & 15, hi = lane >> 4;
        int k = kk * 32 + ((e < 8) ? (hi * 8 + e) : (16 + hi * 8 + e - 8));
        int col = n * 16 + l;                           // 0..511 in [Wg|Wf|Wh|Wt]
        const float* W = (col < 128) ? Wg : (col < 256) ? Wf : (col < 384) ? Wh : Wt;
        Hp[q] = (__bf16)W[k * 128 + (col & 127)];
    }
}

// ---------------------------------------------------------------------------
// K1: z = [x|h] @ W01  via v_wmma_f32_16x16x32_bf16, plus per-column sum/sumsq
// Each wave: 16 rows x 128 cols (8 N-tiles, K=256 in 8 steps). 8 waves/block.
// ---------------------------------------------------------------------------
__global__ __launch_bounds__(256) void cfc_backbone_kernel(
    const float* __restrict__ x, const float* __restrict__ h,
    const __bf16* __restrict__ W01p, float* __restrict__ z,
    float* __restrict__ gstats) {
    __shared__ float lsum[128];
    __shared__ float lsq[128];
    int tid = threadIdx.x;
    if (tid < 128) { lsum[tid] = 0.f; lsq[tid] = 0.f; }
    __syncthreads();

    int wave = tid >> 5, lane = tid & 31;
    int l = lane & 15, hi = lane >> 4;
    int rowbase = blockIdx.x * 128 + wave * 16;
    int arow = rowbase + l;

    v8f acc[8] = {};

#pragma unroll
    for (int kk = 0; kk < 8; ++kk) {
        const int kb = kk * 32;
        const float* src = (kk < 4) ? (x + (size_t)arow * 128 + kb)
                                    : (h + (size_t)arow * 128 + (kb - 128));
        f32x4 a0 = *(const f32x4*)(src + hi * 8);
        f32x4 a1 = *(const f32x4*)(src + hi * 8 + 4);
        f32x4 a2 = *(const f32x4*)(src + 16 + hi * 8);
        f32x4 a3 = *(const f32x4*)(src + 16 + hi * 8 + 4);
        v16bf afrag;
#pragma unroll
        for (int e = 0; e < 4; ++e) {
            afrag[e]      = (__bf16)a0[e];
            afrag[e + 4]  = (__bf16)a1[e];
            afrag[e + 8]  = (__bf16)a2[e];
            afrag[e + 12] = (__bf16)a3[e];
        }
#pragma unroll
        for (int n = 0; n < 8; ++n) {
            v16bf bfrag = *(const v16bf*)(W01p + (((kk * 8 + n) * 32 + lane) << 4));
            acc[n] = __builtin_amdgcn_wmma_f32_16x16x32_bf16(
                false, afrag, false, bfrag, (short)0, acc[n], false, false);
        }
    }

#pragma unroll
    for (int n = 0; n < 8; ++n) {
        float s = 0.f, q = 0.f;
#pragma unroll
        for (int v = 0; v < 8; ++v) {
            float val = acc[n][v];
            z[(size_t)(rowbase + hi * 8 + v) * 128 + n * 16 + l] = val;
            s += val; q += val * val;
        }
        atomicAdd(&lsum[n * 16 + l], s);   // ds_add_f32
        atomicAdd(&lsq[n * 16 + l], q);
    }
    __syncthreads();
    if (tid < 128) {
        atomicAdd(&gstats[tid], lsum[tid]);          // global_atomic_add_f32
        atomicAdd(&gstats[128 + tid], lsq[tid]);
    }
}

// ---------------------------------------------------------------------------
// K2: fold BN stats into per-channel scale/shift
// ---------------------------------------------------------------------------
__global__ void cfc_stats_kernel(const float* __restrict__ gstats,
                                 const float* __restrict__ gamma,
                                 const float* __restrict__ beta,
                                 float* __restrict__ ss) {
    int j = threadIdx.x;
    const float inv = 1.f / (float)B_ROWS;
    float mean = gstats[j] * inv;
    float var  = gstats[128 + j] * inv - mean * mean;
    float sc   = gamma[j] * rsqrtf(var + EPS);
    ss[j]       = sc;
    ss[128 + j] = beta[j] - mean * sc;
}

// ---------------------------------------------------------------------------
// K3: BN + LeakyReLU + 4 fused head GEMMs + CfC gating.
// Reads z rows from d_out, writes the final result over the same rows.
// Per wave: 16 rows; A-fragments (K=128, bf16) built once, reused for all
// 4 heads x 4 column-groups.
// ---------------------------------------------------------------------------
__global__ __launch_bounds__(256) void cfc_heads_kernel(
    const float* __restrict__ zin, const __bf16* __restrict__ Hp,
    const float* __restrict__ ss, const float* __restrict__ tvec,
    const float* __restrict__ bg, const float* __restrict__ bf_,
    const float* __restrict__ bh, const float* __restrict__ bt,
    float* __restrict__ out) {
    int tid = threadIdx.x;
    int wave = tid >> 5, lane = tid & 31;
    int l = lane & 15, hi = lane >> 4;
    int rowbase = blockIdx.x * 128 + wave * 16;
    int arow = rowbase + l;

    v16bf afrag[4];
#pragma unroll
    for (int kk = 0; kk < 4; ++kk) {
        const int kb = kk * 32;
        const float* src = zin + (size_t)arow * 128 + kb;
        const int k0 = hi * 8, k1 = 16 + hi * 8;
        f32x4 z0 = *(const f32x4*)(src + k0);
        f32x4 z1 = *(const f32x4*)(src + k0 + 4);
        f32x4 z2 = *(const f32x4*)(src + k1);
        f32x4 z3 = *(const f32x4*)(src + k1 + 4);
        f32x4 c0 = *(const f32x4*)(ss + kb + k0);
        f32x4 c1 = *(const f32x4*)(ss + kb + k0 + 4);
        f32x4 c2 = *(const f32x4*)(ss + kb + k1);
        f32x4 c3 = *(const f32x4*)(ss + kb + k1 + 4);
        f32x4 s0 = *(const f32x4*)(ss + 128 + kb + k0);
        f32x4 s1 = *(const f32x4*)(ss + 128 + kb + k0 + 4);
        f32x4 s2 = *(const f32x4*)(ss + 128 + kb + k1);
        f32x4 s3 = *(const f32x4*)(ss + 128 + kb + k1 + 4);
#pragma unroll
        for (int e = 0; e < 4; ++e) {
            float u;
            u = z0[e] * c0[e] + s0[e]; u = (u > 0.f) ? u : SLOPE * u; afrag[kk][e]      = (__bf16)u;
            u = z1[e] * c1[e] + s1[e]; u = (u > 0.f) ? u : SLOPE * u; afrag[kk][e + 4]  = (__bf16)u;
            u = z2[e] * c2[e] + s2[e]; u = (u > 0.f) ? u : SLOPE * u; afrag[kk][e + 8]  = (__bf16)u;
            u = z3[e] * c3[e] + s3[e]; u = (u > 0.f) ? u : SLOPE * u; afrag[kk][e + 12] = (__bf16)u;
        }
    }

    float tvals[8];
#pragma unroll
    for (int v = 0; v < 8; ++v) tvals[v] = tvec[rowbase + hi * 8 + v];

#pragma unroll
    for (int cg = 0; cg < 4; ++cg) {
        v8f acc[4][2] = {};
#pragma unroll
        for (int hd = 0; hd < 4; ++hd) {
#pragma unroll
            for (int tt = 0; tt < 2; ++tt) {
                const int n = hd * 8 + cg * 2 + tt;
#pragma unroll
                for (int kk = 0; kk < 4; ++kk) {
                    v16bf bfrag = *(const v16bf*)(Hp + (((kk * 32 + n) * 32 + lane) << 4));
                    acc[hd][tt] = __builtin_amdgcn_wmma_f32_16x16x32_bf16(
                        false, afrag[kk], false, bfrag, (short)0, acc[hd][tt], false, false);
                }
            }
        }
#pragma unroll
        for (int tt = 0; tt < 2; ++tt) {
            const int col = cg * 32 + tt * 16 + l;
            float vg = bg[col], vf = bf_[col], vh = bh[col], vt = bt[col];
#pragma unroll
            for (int v = 0; v < 8; ++v) {
                float g  = tanhf(acc[0][tt][v] + vg);
                float fv = acc[1][tt][v] + vf;
                float hhv= tanhf(acc[2][tt][v] + vh);
                float ta = acc[3][tt][v] + vt;
                float sg = 1.f / (1.f + __expf(-(ta + fv) * tvals[v]));
                out[(size_t)(rowbase + hi * 8 + v) * 128 + col] = hhv * (1.f - sg) + g * sg;
            }
        }
    }
}

// ---------------------------------------------------------------------------
extern "C" void kernel_launch(void* const* d_in, const int* in_sizes, int n_in,
                              void* d_out, int out_size, void* d_ws, size_t ws_size,
                              hipStream_t stream) {
    const float* x     = (const float*)d_in[0];
    const float* h     = (const float*)d_in[1];
    const float* t     = (const float*)d_in[2];
    const float* W0    = (const float*)d_in[3];
    const float* W1    = (const float*)d_in[4];
    const float* gamma = (const float*)d_in[5];
    const float* beta  = (const float*)d_in[6];
    const float* Wg = (const float*)d_in[7];  const float* bg = (const float*)d_in[8];
    const float* Wf = (const float*)d_in[9];  const float* bf = (const float*)d_in[10];
    const float* Wh = (const float*)d_in[11]; const float* bh = (const float*)d_in[12];
    const float* Wt = (const float*)d_in[13]; const float* bt = (const float*)d_in[14];
    float* out = (float*)d_out;

    char* ws = (char*)d_ws;
    float*  W01f  = (float*)ws;                 // 256*128*4   = 131072 B
    __bf16* W01p  = (__bf16*)(ws + 131072);     // 256*128*2   =  65536 B
    __bf16* Hp    = (__bf16*)(ws + 196608);     // 128*512*2   = 131072 B
    float*  stats = (float*)(ws + 327680);      // sum[128] sumsq[128] scale[128] shift[128]

    cfc_w01_kernel     <<<128,  256, 0, stream>>>(W0, W1, W01f);
    cfc_pack_kernel    <<<384,  256, 0, stream>>>(W01f, Wg, Wf, Wh, Wt, W01p, Hp, stats);
    cfc_backbone_kernel<<<1024, 256, 0, stream>>>(x, h, W01p, out, stats);
    cfc_stats_kernel   <<<1,    128, 0, stream>>>(stats, gamma, beta, stats + 256);
    cfc_heads_kernel   <<<1024, 256, 0, stream>>>(out, Hp, stats + 256, t,
                                                  bg, bf, bh, bt, out);
}
